// SiameseEncoder_4715874091353
// MI455X (gfx1250) — matrix operand order repaired and checked by reading, not compiled, not run on hardware
//
#include <hip/hip_runtime.h>
#include <hip/hip_bf16.h>

#define DEV_INLINE __device__ __forceinline__

typedef __attribute__((ext_vector_type(16))) _Float16 v16h;
typedef __attribute__((ext_vector_type(8)))  float    v8f;

namespace vit {
constexpr int B    = 16;
constexpr int T    = 577;   // 24*24 patches + cls
constexpr int D    = 384;
constexpr int H    = 6;
constexpr int HD   = 64;
constexpr int QKVN = 1152;
constexpr int MLPH = 1536;
constexpr int ROWS = B * T;     // 9232
constexpr int NPAT = 576;
constexpr int MPAT = B * NPAT;  // 9216
constexpr int PK   = 256;       // 16x16 patch pixels
constexpr int TKP  = 640;       // key-dim pad (mult of 64) for P*V
constexpr float EPS = 1e-5f;
}

// ---------------------------------------------------------------------------
// Batched NT GEMM: C[M,N] = A[M,K] (f16, row-major) x B[N,K] (f16, row-major)
// Both operands K-major -> all LDS fragment reads are contiguous b128 loads.
// 64x64 output tile, 64-wide K tile (8 WMMAs per barrier pair). 4 waves.
// flags: 1=+bias[n], 2=exact GELU, 4=store f16 (else f32)
// Per-batch z: A += z*sA, B += z*sB, C += (z/zdiv)*sC + (z%zdiv)*sC2
// Requires K % 64 == 0; M and N row ranges are guarded (zero fill).
// ---------------------------------------------------------------------------
__global__ __launch_bounds__(128) void gemm_f16_wmma(
    const _Float16* __restrict__ A, const _Float16* __restrict__ Bm,
    float* __restrict__ Cf, _Float16* __restrict__ Ch,
    const float* __restrict__ bias,
    int M, int N, int K, int lda, int ldb, int ldc,
    long long sA, long long sB, long long sC, long long sC2, int zdiv,
    int flags)
{
  __shared__ _Float16 As[64][72];   // 64 rows (M) x 64 K (+8 pad)
  __shared__ _Float16 Bs[64][72];   // 64 rows (N) x 64 K (+8 pad)

  const int z = blockIdx.z;
  A  += (long long)z * sA;
  Bm += (long long)z * sB;
  const long long coff = (long long)(z / zdiv) * sC + (long long)(z % zdiv) * sC2;

  const int tid  = threadIdx.x;
  const int wave = tid >> 5;
  const int lane = tid & 31;
  const int l15  = lane & 15;
  const int hi   = lane >> 4;

  const int m0 = blockIdx.y * 64;
  const int n0 = blockIdx.x * 64;

  union F8  { v8f  v; float f[8]; };
  union F16 { v16h v; uint4 u4[2]; _Float16 h[16]; };
  F8 acc[4];
#pragma unroll
  for (int c = 0; c < 4; ++c)
#pragma unroll
    for (int r = 0; r < 8; ++r) acc[c].f[r] = 0.f;

  const int sr = tid >> 1;         // staging row 0..63 (used for both tiles)
  const int sc = (tid & 1) * 32;   // staging K chunk {0,32}

  for (int k0 = 0; k0 < K; k0 += 64) {
    // ---- stage A tile rows (M-guarded, zero fill) ----
    {
      const int gm = m0 + sr;
      uint4 d0 = make_uint4(0u,0u,0u,0u), d1 = d0, d2 = d0, d3 = d0;
      if (gm < M) {
        const _Float16* src = A + (long long)gm * lda + (k0 + sc);
        d0 = *(const uint4*)(src);
        d1 = *(const uint4*)(src + 8);
        d2 = *(const uint4*)(src + 16);
        d3 = *(const uint4*)(src + 24);
      }
      *(uint4*)&As[sr][sc]      = d0;
      *(uint4*)&As[sr][sc + 8]  = d1;
      *(uint4*)&As[sr][sc + 16] = d2;
      *(uint4*)&As[sr][sc + 24] = d3;
    }
    // ---- stage B tile rows (N-guarded, zero fill) ----
    {
      const int gn = n0 + sr;
      uint4 d0 = make_uint4(0u,0u,0u,0u), d1 = d0, d2 = d0, d3 = d0;
      if (gn < N) {
        const _Float16* src = Bm + (long long)gn * ldb + (k0 + sc);
        d0 = *(const uint4*)(src);
        d1 = *(const uint4*)(src + 8);
        d2 = *(const uint4*)(src + 16);
        d3 = *(const uint4*)(src + 24);
      }
      *(uint4*)&Bs[sr][sc]      = d0;
      *(uint4*)&Bs[sr][sc + 8]  = d1;
      *(uint4*)&Bs[sr][sc + 16] = d2;
      *(uint4*)&Bs[sr][sc + 24] = d3;
    }
    if (k0 + 64 < K) {   // speculative prefetch of next K tile -> global_prefetch_b8
      __builtin_prefetch(A  + (long long)(m0 + sr) * lda + (k0 + 64 + sc), 0, 1);
      __builtin_prefetch(Bm + (long long)(n0 + sr) * ldb + (k0 + 64 + sc), 0, 1);
    }
    __syncthreads();

#pragma unroll
    for (int kk = 0; kk < 64; kk += 32) {
      // ---- A fragment: two contiguous 8-half runs (K=kk+hi*8+e, +16) ----
      F16 af;
      const int am = wave * 16 + l15;
      af.u4[0] = *(const uint4*)&As[am][kk + hi * 8];
      af.u4[1] = *(const uint4*)&As[am][kk + hi * 8 + 16];
      // ---- preload all four B fragments, then issue WMMAs back-to-back ----
      F16 bf[4];
#pragma unroll
      for (int c = 0; c < 4; ++c) {
        const int bn = c * 16 + l15;
        bf[c].u4[0] = *(const uint4*)&Bs[bn][kk + hi * 8];
        bf[c].u4[1] = *(const uint4*)&Bs[bn][kk + hi * 8 + 16];
      }
#pragma unroll
      for (int c = 0; c < 4; ++c) {
        acc[c].v = __builtin_amdgcn_wmma_f32_16x16x32_f16(
            false, af.v, false, bf[c].v, (short)0, acc[c].v, false, false);
      }
    }
    __syncthreads();
  }

  // ---- epilogue: C layout VGPR r -> row r + 8*hi, lane l15 -> col ----
#pragma unroll
  for (int c = 0; c < 4; ++c) {
#pragma unroll
    for (int r = 0; r < 8; ++r) {
      const int gm = m0 + wave * 16 + hi * 8 + r;
      const int gn = n0 + c * 16 + l15;
      if (gm < M && gn < N) {
        float v = acc[c].f[r];
        if (flags & 1) v += bias[gn];
        if (flags & 2) v = 0.5f * v * (1.0f + erff(v * 0.70710678118654752f));
        const long long o = coff + (long long)gm * ldc + gn;
        if (flags & 4) Ch[o] = (_Float16)v;
        else           Cf[o] = v;
      }
    }
  }
}

// ---------------------------------------------------------------------------
// 384-thread block reduction (one element per thread over D=384)
// ---------------------------------------------------------------------------
DEV_INLINE float breduce384(float v, float* red) {
  const int t = threadIdx.x;
  red[t] = v;
  __syncthreads();
  if (t < 128) red[t] += red[t + 128] + red[t + 256];
  __syncthreads();
  for (int s = 64; s > 0; s >>= 1) {
    if (t < s) red[t] += red[t + s];
    __syncthreads();
  }
  const float r = red[0];
  __syncthreads();
  return r;
}

// LayerNorm row -> f16 (GEMM operand)
__global__ __launch_bounds__(384) void ln_rows_f16(
    const float* __restrict__ x, const float* __restrict__ w,
    const float* __restrict__ b, _Float16* __restrict__ y)
{
  __shared__ float red[384];
  const int d = threadIdx.x;
  const long long idx = (long long)blockIdx.x * vit::D + d;
  const float v  = x[idx];
  const float m  = breduce384(v, red) * (1.0f / vit::D);
  const float dv = v - m;
  const float va = breduce384(dv * dv, red) * (1.0f / vit::D);
  y[idx] = (_Float16)(dv * rsqrtf(va + vit::EPS) * w[d] + b[d]);
}

// Final LN on the cls token only -> fp32 output [B, D]
__global__ __launch_bounds__(384) void final_ln(
    const float* __restrict__ hidden, const float* __restrict__ w,
    const float* __restrict__ b, float* __restrict__ out)
{
  __shared__ float red[384];
  const int d = threadIdx.x;
  const long long idx = (long long)blockIdx.x * vit::T * vit::D + d;  // t = 0
  const float v  = hidden[idx];
  const float m  = breduce384(v, red) * (1.0f / vit::D);
  const float dv = v - m;
  const float va = breduce384(dv * dv, red) * (1.0f / vit::D);
  out[blockIdx.x * vit::D + d] = dv * rsqrtf(va + vit::EPS) * w[d] + b[d];
}

// block_res: LN each stacked tensor, logits = K.pw, softmax over stack, mix.
// One block per (b,t) row; thread d owns element d. n = nstack + 1 (<= 7).
__global__ __launch_bounds__(384) void block_res_kernel(
    const float* __restrict__ stack, int nstack,
    const float* __restrict__ extra,
    const float* __restrict__ nw, const float* __restrict__ nb,
    const float* __restrict__ pw, float* __restrict__ out)
{
  __shared__ float red[384];
  __shared__ float logits[8];
  const int d = threadIdx.x;
  const long long idx = (long long)blockIdx.x * vit::D + d;
  const long long sstride = (long long)vit::ROWS * vit::D;
  const int n = nstack + 1;
  float vals[7];
  for (int s = 0; s < n; ++s) {
    const float* src = (s < nstack) ? (stack + s * sstride) : extra;
    const float v = src[idx];
    vals[s] = v;
    const float m  = breduce384(v, red) * (1.0f / vit::D);
    const float dv = v - m;
    const float va = breduce384(dv * dv, red) * (1.0f / vit::D);
    const float kn = dv * rsqrtf(va + vit::EPS) * nw[d] + nb[d];
    const float lg = breduce384(kn * pw[d], red);
    if (d == 0) logits[s] = lg;
  }
  __syncthreads();
  float mx = -1e30f;
  for (int s = 0; s < n; ++s) mx = fmaxf(mx, logits[s]);
  float den = 0.f;
  for (int s = 0; s < n; ++s) den += expf(logits[s] - mx);
  float o = 0.f;
  for (int s = 0; s < n; ++s) o += vals[s] * (expf(logits[s] - mx) / den);
  out[idx] = o;
}

// RoPE + head split/pack:
//   qkv f32 [B,T,3*D] -> qh, kh f16 [B,H,TKP,HD] (row-major, K-major for NT GEMM)
//                        vT    f16 [B,H,HD,TKP] (V transposed: PV GEMM B operand)
__global__ __launch_bounds__(64) void rope_pack(
    const float* __restrict__ qkv, _Float16* __restrict__ qh,
    _Float16* __restrict__ kh, _Float16* __restrict__ vT)
{
  const int t  = blockIdx.x;          // 0..TKP-1
  const int bh = blockIdx.y;          // b*H + h
  const int b  = bh / vit::H;
  const int h  = bh % vit::H;
  const int d  = threadIdx.x;         // 0..63
  const long long qi = ((long long)bh * vit::TKP + t) * vit::HD + d;
  const long long vi = ((long long)bh * vit::HD + d) * vit::TKP + t;
  if (t < vit::T) {
    const long long base = ((long long)(b * vit::T + t)) * vit::QKVN + h * vit::HD;
    const float q = qkv[base + d];
    const float k = qkv[base + vit::D + d];
    const float v = qkv[base + 2 * vit::D + d];
    const int j = d & 31;                                    // emb = concat(fr, fr)
    const float inv = expf(-(float)(2 * j) * (1.0f / 64.0f) * 9.210340371976184f); // ln(10000)
    const float fr = (float)t * inv;
    const float cs = cosf(fr), sn = sinf(fr);
    const float qp = (d < 32) ? -qkv[base + d + 32]          : qkv[base + d - 32];
    const float kp = (d < 32) ? -qkv[base + vit::D + d + 32] : qkv[base + vit::D + d - 32];
    qh[qi] = (_Float16)(q * cs + qp * sn);
    kh[qi] = (_Float16)(k * cs + kp * sn);
    vT[vi] = (_Float16)v;
  } else {
    qh[qi] = (_Float16)0.f;
    kh[qi] = (_Float16)0.f;
    vT[vi] = (_Float16)0.f;
  }
}

// Row softmax of scaled scores; writes f16 probs zero-padded to TKP columns.
__global__ __launch_bounds__(128) void softmax_rows(
    const float* __restrict__ S, _Float16* __restrict__ P, float scale)
{
  __shared__ float red[128];
  __shared__ float mval, sval;
  const long long row = blockIdx.x;
  const float* s = S + row * vit::T;
  _Float16*    p = P + row * vit::TKP;
  const int t = threadIdx.x;
  float mx = -1e30f;
  for (int k = t; k < vit::T; k += 128) mx = fmaxf(mx, s[k] * scale);
  red[t] = mx; __syncthreads();
  for (int st = 64; st > 0; st >>= 1) { if (t < st) red[t] = fmaxf(red[t], red[t + st]); __syncthreads(); }
  if (t == 0) mval = red[0];
  __syncthreads();
  float sum = 0.f;
  for (int k = t; k < vit::T; k += 128) sum += expf(s[k] * scale - mval);
  red[t] = sum; __syncthreads();
  for (int st = 64; st > 0; st >>= 1) { if (t < st) red[t] += red[t + st]; __syncthreads(); }
  if (t == 0) sval = red[0];
  __syncthreads();
  const float inv = 1.0f / sval;
  for (int k = t; k < vit::TKP; k += 128)
    p[k] = (k < vit::T) ? (_Float16)(expf(s[k] * scale - mval) * inv) : (_Float16)0.f;
}

// Gather 16x16 patches -> A f16 [MPAT, PK]
__global__ void pack_patches(const float* __restrict__ x, _Float16* __restrict__ ap)
{
  const int idx = blockIdx.x * blockDim.x + threadIdx.x;
  if (idx >= vit::MPAT * vit::PK) return;
  const int r = idx >> 8, k = idx & 255;
  const int b = r / vit::NPAT, pp = r % vit::NPAT;
  const int py = pp / 24, px = pp % 24;
  const int iy = k >> 4, ix = k & 15;
  ap[idx] = (_Float16)x[((long long)b * 384 + (py * 16 + iy)) * 384 + (px * 16 + ix)];
}

// Straight f32 -> f16 cast (weights are [out,in] = [N,K] already: NT GEMM needs no transpose)
__global__ void cast_f16(const float* __restrict__ w, _Float16* __restrict__ o, int n)
{
  const int idx = blockIdx.x * blockDim.x + threadIdx.x;
  if (idx < n) o[idx] = (_Float16)w[idx];
}

// h[b,0,:]=cls ; h[b,1+p,:]=patch tokens
__global__ void assemble_tokens(const float* __restrict__ pt, const float* __restrict__ cls,
                                float* __restrict__ h)
{
  const int idx = blockIdx.x * blockDim.x + threadIdx.x;
  if (idx >= vit::ROWS * vit::D) return;
  const int row = idx / vit::D, d = idx % vit::D;
  const int b = row / vit::T, t = row % vit::T;
  h[idx] = (t == 0) ? cls[d]
                    : pt[((long long)(b * vit::NPAT + (t - 1))) * vit::D + d];
}

__global__ void make_partial(float* __restrict__ partial, const float* __restrict__ hidden,
                             const float* __restrict__ a, int add_hidden)
{
  const int idx = blockIdx.x * blockDim.x + threadIdx.x;
  if (idx >= vit::ROWS * vit::D) return;
  partial[idx] = a[idx] + (add_hidden ? hidden[idx] : 0.f);
}

__global__ void add_into_hidden(float* __restrict__ hidden, const float* __restrict__ partial,
                                const float* __restrict__ m)
{
  const int idx = blockIdx.x * blockDim.x + threadIdx.x;
  if (idx >= vit::ROWS * vit::D) return;
  hidden[idx] = partial[idx] + m[idx];
}

// ---------------------------------------------------------------------------
extern "C" void kernel_launch(void* const* d_in, const int* in_sizes, int n_in,
                              void* d_out, int out_size, void* d_ws, size_t ws_size,
                              hipStream_t stream)
{
  using namespace vit;
  (void)out_size; (void)ws_size; (void)n_in;

  struct Layer {
    const float *ar_nw,*ar_nb,*ar_pw,*mr_nw,*mr_nb,*mr_pw,*an_w,*an_b,*mn_w,*mn_b,
                *qkv_w,*pr_w,*pr_b,*fc1_w,*fc1_b,*fc2_w,*fc2_b;
  } L[10];
  const float *X, *PWc, *CLS, *NWf, *NBf;
  auto F = [&](int i) { return (const float*)d_in[i]; };

  if (in_sizes[0] == 16 * 1 * 384 * 384) {
    // dict-insertion order: x, patch_w, cls, norm_w, norm_b, layers(ar_nw..fc2_b)
    X = F(0); PWc = F(1); CLS = F(2); NWf = F(3); NBf = F(4);
    for (int l = 0; l < 10; ++l) {
      const int b0 = 5 + l * 17;
      L[l] = { F(b0+0),F(b0+1),F(b0+2),F(b0+3),F(b0+4),F(b0+5),F(b0+6),F(b0+7),F(b0+8),
               F(b0+9),F(b0+10),F(b0+11),F(b0+12),F(b0+13),F(b0+14),F(b0+15),F(b0+16) };
    }
  } else {
    // sorted-key pytree order: cls, layers(sorted keys), norm_b, norm_w, patch_w, x
    CLS = F(0);
    for (int l = 0; l < 10; ++l) {
      const int b0 = 1 + l * 17;
      L[l].an_b = F(b0+0);  L[l].an_w = F(b0+1);  L[l].ar_nb = F(b0+2);  L[l].ar_nw = F(b0+3);
      L[l].ar_pw = F(b0+4); L[l].fc1_b = F(b0+5); L[l].fc1_w = F(b0+6);  L[l].fc2_b = F(b0+7);
      L[l].fc2_w = F(b0+8); L[l].mn_b = F(b0+9);  L[l].mn_w = F(b0+10);  L[l].mr_nb = F(b0+11);
      L[l].mr_nw = F(b0+12);L[l].mr_pw = F(b0+13);L[l].pr_b = F(b0+14);  L[l].pr_w = F(b0+15);
      L[l].qkv_w = F(b0+16);
    }
    NBf = F(171); NWf = F(172); PWc = F(173); X = F(174);
  }

  // ---- workspace carve ----
  char* base = (char*)d_ws;
  size_t off = 0;
  auto alloc = [&](size_t bytes) -> void* {
    void* p = base + off;
    off += (bytes + 255) & ~(size_t)255;
    return p;
  };
  float*    stack   = (float*)alloc(sizeof(float) * 6ull * ROWS * D);
  float*    hidden  = (float*)alloc(sizeof(float) * (size_t)ROWS * D);
  float*    partial = (float*)alloc(sizeof(float) * (size_t)ROWS * D);
  float*    hmix    = (float*)alloc(sizeof(float) * (size_t)ROWS * D);
  float*    abuf    = (float*)alloc(sizeof(float) * (size_t)ROWS * D);
  float*    mbuf    = (float*)alloc(sizeof(float) * (size_t)ROWS * D);
  float*    qkv     = (float*)alloc(sizeof(float) * (size_t)ROWS * QKVN);
  float*    scores  = (float*)alloc(sizeof(float) * (size_t)B * H * T * T);
  _Float16* ln16    = (_Float16*)alloc(2ull * ROWS * D);
  _Float16* qh      = (_Float16*)alloc(2ull * B * H * TKP * HD);
  _Float16* kh      = (_Float16*)alloc(2ull * B * H * TKP * HD);
  _Float16* vT      = (_Float16*)alloc(2ull * B * H * HD * TKP);
  _Float16* p16     = (_Float16*)alloc(2ull * B * H * T * TKP);
  _Float16* am16    = (_Float16*)alloc(2ull * ROWS * D);
  _Float16* mlp16   = (_Float16*)alloc(2ull * ROWS * MLPH);
  _Float16* wq16    = (_Float16*)alloc(2ull * QKVN * D);
  _Float16* wp16    = (_Float16*)alloc(2ull * D * D);
  _Float16* w116    = (_Float16*)alloc(2ull * MLPH * D);
  _Float16* w216    = (_Float16*)alloc(2ull * D * MLPH);
  _Float16* wpa16   = (_Float16*)alloc(2ull * D * PK);
  _Float16* apat    = (_Float16*)alloc(2ull * MPAT * PK);

  auto gemm = [&](const _Float16* A, int lda, long long sA,
                  const _Float16* Bmat, int ldb, long long sB,
                  float* Cf, _Float16* Ch, int ldc, long long sC, long long sC2, int zdiv,
                  int M, int N, int K, const float* bias, int flags, int batch) {
    dim3 g((unsigned)((N + 63) / 64), (unsigned)((M + 63) / 64), (unsigned)batch);
    gemm_f16_wmma<<<g, dim3(128), 0, stream>>>(A, Bmat, Cf, Ch, bias, M, N, K,
                                               lda, ldb, ldc, sA, sB, sC, sC2, zdiv, flags);
  };

  // ---- patch embedding ----
  pack_patches<<<(MPAT * PK + 255) / 256, 256, 0, stream>>>(X, apat);
  cast_f16<<<(D * PK + 255) / 256, 256, 0, stream>>>(PWc, wpa16, D * PK);
  gemm(apat, PK, 0, wpa16, PK, 0, abuf, nullptr, D, 0, 0, 1, MPAT, D, PK, nullptr, 0, 1);
  assemble_tokens<<<(ROWS * D + 255) / 256, 256, 0, stream>>>(abuf, CLS, stack /*slot 0*/);
  hipMemcpyAsync(hidden, stack, sizeof(float) * (size_t)ROWS * D,
                 hipMemcpyDeviceToDevice, stream);

  const size_t slotBytes = sizeof(float) * (size_t)ROWS * D;
  int nb = 1;
  for (int l = 0; l < 10; ++l) {
    const Layer& lp = L[l];
    // f16 casts of this layer's weights ([N,K] layout as stored — no transpose)
    cast_f16<<<(QKVN * D + 255) / 256, 256, 0, stream>>>(lp.qkv_w, wq16, QKVN * D);
    cast_f16<<<(D * D    + 255) / 256, 256, 0, stream>>>(lp.pr_w,  wp16, D * D);
    cast_f16<<<(MLPH * D + 255) / 256, 256, 0, stream>>>(lp.fc1_w, w116, MLPH * D);
    cast_f16<<<(D * MLPH + 255) / 256, 256, 0, stream>>>(lp.fc2_w, w216, D * MLPH);

    // hmix = block_res(blocks + [hidden])
    block_res_kernel<<<ROWS, 384, 0, stream>>>(stack, nb, hidden,
                                               lp.ar_nw, lp.ar_nb, lp.ar_pw, hmix);
    const bool append = (l % 2 == 0);
    if (append)
      hipMemcpyAsync(stack + (size_t)nb * ROWS * D, hidden, slotBytes,
                     hipMemcpyDeviceToDevice, stream);
    const int nb2 = append ? nb + 1 : nb;

    // attention
    ln_rows_f16<<<ROWS, 384, 0, stream>>>(hmix, lp.an_w, lp.an_b, ln16);
    gemm(ln16, D, 0, wq16, D, 0, qkv, nullptr, QKVN, 0, 0, 1,
         ROWS, QKVN, D, nullptr, 0, 1);
    rope_pack<<<dim3(TKP, B * H), 64, 0, stream>>>(qkv, qh, kh, vT);
    gemm(qh, HD, (long long)TKP * HD, kh, HD, (long long)TKP * HD,
         scores, nullptr, T, (long long)T * T, 0, 1,
         T, T, HD, nullptr, 0, B * H);                         // S = q k^T
    softmax_rows<<<B * H * T, 128, 0, stream>>>(scores, p16, 0.125f);
    gemm(p16, TKP, (long long)T * TKP, vT, TKP, (long long)HD * TKP,
         nullptr, am16, D, (long long)T * D, (long long)HD, H,
         T, HD, TKP, nullptr, 4, B * H);                       // head-merged f16 out
    gemm(am16, D, 0, wp16, D, 0, abuf, nullptr, D, 0, 0, 1,
         ROWS, D, D, lp.pr_b, 1, 1);
    make_partial<<<(ROWS * D + 255) / 256, 256, 0, stream>>>(partial, hidden, abuf,
                                                             append ? 0 : 1);

    // MLP branch
    block_res_kernel<<<ROWS, 384, 0, stream>>>(stack, nb2, partial,
                                               lp.mr_nw, lp.mr_nb, lp.mr_pw, hmix);
    ln_rows_f16<<<ROWS, 384, 0, stream>>>(hmix, lp.mn_w, lp.mn_b, ln16);
    gemm(ln16, D, 0, w116, D, 0, nullptr, mlp16, MLPH, 0, 0, 1,
         ROWS, MLPH, D, lp.fc1_b, 1 | 2 | 4, 1);               // bias + GELU -> f16
    gemm(mlp16, MLPH, 0, w216, MLPH, 0, mbuf, nullptr, D, 0, 0, 1,
         ROWS, D, MLPH, lp.fc2_b, 1, 1);
    add_into_hidden<<<(ROWS * D + 255) / 256, 256, 0, stream>>>(hidden, partial, mbuf);
    nb = nb2;
  }

  final_ln<<<B, 384, 0, stream>>>(hidden, NWf, NBf, (float*)d_out);
}
